// Dy_UpSample_33097017983646
// MI455X (gfx1250) — compile-verified, matrix operand order checked by reading, not run on hardware
//
#include <hip/hip_runtime.h>

typedef __attribute__((ext_vector_type(16))) _Float16 v16h;
typedef __attribute__((ext_vector_type(8)))  float    v8f;
typedef __attribute__((ext_vector_type(4)))  uint32_t v4u;
typedef __attribute__((ext_vector_type(8)))  uint32_t v8u;

#define B_    16
#define C_    256
#define H_    64
#define W_    64
#define HW_   4096
#define HO_   128
#define WO_   128

// Workspace layout:
//   [0, 8 MB)            : off_ps  float (B, 4, HO, WO, 2)  — modulated, pixel-shuffled offsets
//   [8 MB, 8 MB + 32 KB) : Wcat    f16   (64, 256)          — [W_off ; block-diag W_scope]

// ---------------------------------------------------------------------------
// Kernel 0: pack fused weight matrix (rows 0-31 = W_off, rows 32-63 = scope
// weights embedded block-diagonally so one dense GEMM computes both convs).
// ---------------------------------------------------------------------------
__global__ void prep_weights(const float* __restrict__ w_off,
                             const float* __restrict__ w_scope,
                             _Float16* __restrict__ Wcat) {
    int idx = blockIdx.x * blockDim.x + threadIdx.x;   // 0 .. 16383
    int row = idx >> 8;
    int k   = idx & 255;
    float v;
    if (row < 32) {
        v = w_off[row * 256 + k];
    } else {
        int s = row - 32;               // scope channel 0..31
        int g = s >> 3, o = s & 7;      // group, within-group out channel
        v = ((k >> 6) == g) ? w_scope[(g * 8 + o) * 64 + (k & 63)] : 0.0f;
    }
    Wcat[idx] = (_Float16)v;
}

// ---------------------------------------------------------------------------
// Kernel 1: fused off/scope GEMM via v_wmma_f32_16x16x32_f16, with the 32 KB
// weight matrix staged into LDS once per workgroup via the Tensor Data Mover.
// One wave handles a 16-pixel tile (N=16) and all 64 output rows (M-tiles 0..3,
// rows 0-31 = off, rows 32-63 = sc). K = 256 -> 8 WMMA K-steps.
// Epilogue: off' = sigmoid(sc)*0.5*off, then pixel-shuffle scatter to
// off_ps[b][g][ho][wo][d].
// ---------------------------------------------------------------------------
__global__ void __launch_bounds__(256)
offset_gemm(const float* __restrict__ x,
            const _Float16* __restrict__ Wcat,
            const float* __restrict__ b_off,
            const float* __restrict__ b_scope,
            float* __restrict__ off_ps) {
    __shared__ _Float16 Wlds[64 * 256];      // 32 KB staged weights

    // ---- TDM stage: wave 0 issues one tensor_load_to_lds for the workgroup.
    // D# per ISA §8: 2D tensor, 64 rows x 256 f16, dense (stride0 = 256),
    // tile = whole tensor, data_size = 2 bytes (code 1), type = 2 ("image").
    // NOTE: lds_addr comes from the real pointer — this both fills the
    // descriptor correctly and escapes Wlds so the asm "memory" clobber is
    // understood to write it (otherwise the LDS reads fold to undef).
    const uint32_t lds_off = (uint32_t)(uintptr_t)(&Wlds[0]);
    if (threadIdx.x < 32) {
        union { uint32_t w[4]; v4u v; } g0 = {};
        union { uint32_t w[8]; v8u v; } g1 = {};
        const uint64_t ga = (uint64_t)(uintptr_t)Wcat;
        g0.w[0] = 1u;                                   // count=1 (valid user D#)
        g0.w[1] = lds_off;                              // lds_addr (bytes)
        g0.w[2] = (uint32_t)ga;                         // global_addr[31:0]
        g0.w[3] = (uint32_t)(ga >> 32) | (2u << 30);    // global_addr[56:32] | type=2
        g1.w[0] = (1u << 16);                           // data_size = 2B; mask=0
        g1.w[1] = (256u & 0xFFFFu) << 16;               // tensor_dim0 lo (bits 48..63)
        g1.w[2] = (256u >> 16) | (64u << 16);           // tensor_dim0 hi | tensor_dim1=64
        g1.w[3] = (256u << 16);                         // tile_dim0 = 256 (bits 112..127)
        g1.w[4] = 64u;                                  // tile_dim1 = 64
        g1.w[5] = 256u;                                 // tensor_dim0_stride = 256
        asm volatile("tensor_load_to_lds %0, %1"
                     :: "s"(g0.v), "s"(g1.v) : "memory");
        __builtin_amdgcn_s_wait_tensorcnt((short)0);
    }
    __syncthreads();

    const int lane = threadIdx.x & 31;
    const int wave = (blockIdx.x * blockDim.x + threadIdx.x) >> 5;  // 0..4095
    const int hl   = lane >> 4;     // lane half (K/M split per ISA layout)
    const int l    = lane & 15;

    const int p0  = wave << 4;      // first flat pixel of this 16-pixel tile
    const int img = p0 >> 12;       // / HW_ (tiles never straddle images)
    const int hw0 = p0 & 4095;

    // column l of the B (x) tile: x[img, k, hw0 + l], k-stride = HW_
    const float* xb = x + ((size_t)img * C_) * (size_t)HW_ + hw0 + l;

    v8f acc[4] = {v8f{}, v8f{}, v8f{}, v8f{}};

    for (int kk = 0; kk < 8; ++kk) {
        const int k0 = kk * 32;

        // ---- B fragment: 32x16 f16 tile of x (f32 -> f16 on the fly).
        // lane half 0 holds K = k0+{0..7, 16..23}; half 1 holds +8 of each.
        v16h bf;
        const float* xk = xb + (size_t)(k0 + hl * 8) * HW_;
        #pragma unroll
        for (int j = 0; j < 8; ++j) bf[j] = (_Float16)xk[(size_t)j * HW_];
        const float* xk2 = xk + (size_t)16 * HW_;
        #pragma unroll
        for (int j = 0; j < 8; ++j) bf[8 + j] = (_Float16)xk2[(size_t)j * HW_];

        // ---- A fragments from LDS + 4 WMMAs reusing bf
        #pragma unroll
        for (int t = 0; t < 4; ++t) {
            const _Float16* wr = Wlds + (size_t)(t * 16 + l) * 256 + k0 + hl * 8;
            v16h af;
            #pragma unroll
            for (int j = 0; j < 8; ++j) af[j] = wr[j];
            #pragma unroll
            for (int j = 0; j < 8; ++j) af[8 + j] = wr[16 + j];

            acc[t] = __builtin_amdgcn_wmma_f32_16x16x32_f16(
                false, af, false, bf, (short)0, acc[t], false, false);
        }
    }

    // ---- Epilogue: modulate + pixel-shuffle scatter.
    // C layout: VGPR r, lanes 0-15 -> row 16t+r, lanes 16-31 -> row 16t+r+8.
    const int p  = p0 + l;
    const int hw = p & 4095;
    const int h  = hw >> 6;
    const int w  = hw & 63;

    #pragma unroll
    for (int t2 = 0; t2 < 2; ++t2) {            // off tile t2, sc tile t2+2
        #pragma unroll
        for (int r = 0; r < 8; ++r) {
            const int o = t2 * 16 + r + hl * 8; // off/sc channel 0..31
            float offv = acc[t2][r]     + b_off[o];
            float scv  = acc[t2 + 2][r] + b_scope[o];
            float m = 0.5f * offv * (1.0f / (1.0f + __expf(-scv)));
            // channel o = 8*g + 4*d + 2*ry + rx  (pixel-shuffle + group split)
            const int gp = o >> 3;
            const int d  = (o >> 2) & 1;
            const int ry = (o >> 1) & 1;
            const int rx = o & 1;
            const int ho = 2 * h + ry;
            const int wo = 2 * w + rx;
            off_ps[((((size_t)img * 4 + gp) * HO_ + ho) * WO_ + wo) * 2 + d] = m;
        }
    }
}

// ---------------------------------------------------------------------------
// Kernel 2: bilinear gather (the bandwidth kernel).
// One thread per (b, g, ho, wo); corner indices/weights computed once, then a
// 64-channel loop (group channels are c = i*4 + g). Stores are fully coalesced
// 128B lines per channel; x gathers have strong 2D locality and x (64 MB) fits
// in the 192 MB L2.
// ---------------------------------------------------------------------------
__global__ void __launch_bounds__(256)
sample_kernel(const float* __restrict__ x,
              const float* __restrict__ off_ps,
              float* __restrict__ out) {
    const int idx = blockIdx.x * blockDim.x + threadIdx.x;  // B*4*HO*WO
    const int wo = idx & 127;
    const int ho = (idx >> 7) & 127;
    const int gp = (idx >> 14) & 3;
    const int b  = idx >> 16;

    const float2 og = *(const float2*)(off_ps + (size_t)idx * 2);

    // ix = ((gx+1)*W - 1)/2 with gx = 2*(pos+off)/Wo - 1  ==>  (pos+off)*0.5-0.5
    float ix = ((float)wo + 0.5f + og.x) * 0.5f - 0.5f;
    float iy = ((float)ho + 0.5f + og.y) * 0.5f - 0.5f;
    ix = fminf(fmaxf(ix, 0.0f), (float)(W_ - 1));
    iy = fminf(fmaxf(iy, 0.0f), (float)(H_ - 1));

    const float x0f = floorf(ix), y0f = floorf(iy);
    const float wx = ix - x0f, wy = iy - y0f;
    int xi0 = (int)x0f; xi0 = xi0 < 0 ? 0 : (xi0 > W_ - 1 ? W_ - 1 : xi0);
    int yi0 = (int)y0f; yi0 = yi0 < 0 ? 0 : (yi0 > H_ - 1 ? H_ - 1 : yi0);
    const int xi1 = xi0 + 1 > W_ - 1 ? W_ - 1 : xi0 + 1;
    const int yi1 = yi0 + 1 > H_ - 1 ? H_ - 1 : yi0 + 1;

    const float w00 = (1.0f - wy) * (1.0f - wx);
    const float w01 = (1.0f - wy) * wx;
    const float w10 = wy * (1.0f - wx);
    const float w11 = wy * wx;

    const int o00 = yi0 * W_ + xi0, o01 = yi0 * W_ + xi1;
    const int o10 = yi1 * W_ + xi0, o11 = yi1 * W_ + xi1;

    const size_t xbase  = ((size_t)b * C_ + gp) * (size_t)HW_;        // c = i*4+gp
    const size_t obase  = ((size_t)b * C_ + gp) * (size_t)(HO_ * WO_)
                        + (size_t)ho * WO_ + wo;
    const size_t xcstep = (size_t)4 * HW_;          // channel step in x
    const size_t ocstep = (size_t)4 * (HO_ * WO_);  // channel step in out

    #pragma unroll 4
    for (int i = 0; i < 64; ++i) {
        const float* xc = x + xbase + (size_t)i * xcstep;
        __builtin_prefetch(xc + xcstep + o00, 0, 0);  // next channel's line
        float v = w00 * xc[o00] + w01 * xc[o01]
                + w10 * xc[o10] + w11 * xc[o11];
        out[obase + (size_t)i * ocstep] = v;
    }
}

// ---------------------------------------------------------------------------
extern "C" void kernel_launch(void* const* d_in, const int* in_sizes, int n_in,
                              void* d_out, int out_size, void* d_ws, size_t ws_size,
                              hipStream_t stream) {
    const float* x       = (const float*)d_in[0];
    const float* w_off   = (const float*)d_in[1];
    const float* b_off   = (const float*)d_in[2];
    const float* w_scope = (const float*)d_in[3];
    const float* b_scope = (const float*)d_in[4];
    float* out = (float*)d_out;

    float*    off_ps = (float*)d_ws;                                   // 8 MB
    _Float16* Wcat   = (_Float16*)((char*)d_ws + ((size_t)8 << 20));   // 32 KB

    // 64x256 fused weight matrix
    prep_weights<<<64, 256, 0, stream>>>(w_off, w_scope, Wcat);

    // 4096 pixel-tiles, 8 waves (tiles) per 256-thread block
    offset_gemm<<<512, 256, 0, stream>>>(x, Wcat, b_off, b_scope, off_ps);

    // B*4*HO*WO = 1,048,576 output-pixel threads
    sample_kernel<<<4096, 256, 0, stream>>>(x, off_ps, out);
}
